// InstitutionalTrader_3564822856260
// MI455X (gfx1250) — compile-verified, matrix operand order checked by reading, not run on hardware
//
#include <hip/hip_runtime.h>
#include <hip/hip_bf16.h>

// ---------------------------------------------------------------------------
// Problem constants (from the reference)
// ---------------------------------------------------------------------------
#define T_STEPS 256
#define N_NODES 2000
#define N_EDGES 64000
#define NODE_DIM 256
#define KPI_DIM 32
#define HID 128
#define IN_DIM (HID + KPI_DIM)      // 160

#define M_ROWS (T_STEPS * N_NODES)  // 512000 rows in the big GEMM (divisible by 128)

typedef __attribute__((ext_vector_type(16))) _Float16 v16h;
typedef __attribute__((ext_vector_type(8)))  float    v8f;

// ---------------------------------------------------------------------------
// Kernel 1: per-timestep degree -> dinv = rsqrt(deg), deg includes self loop.
// 2000 counters live in LDS (8 KB), ds_add_f32 atomics.
// ---------------------------------------------------------------------------
__global__ __launch_bounds__(256)
void deg_kernel(const int* __restrict__ ei, float* __restrict__ dinv) {
    __shared__ float cnt[N_NODES];
    const int t   = blockIdx.x;
    const int tid = threadIdx.x;
    const int* dst = ei + (size_t)t * 2 * N_EDGES + N_EDGES;
    for (int i = tid; i < N_NODES; i += 256) cnt[i] = 1.0f;   // self loop
    __syncthreads();
    for (int e = tid; e < N_EDGES; e += 256) atomicAdd(&cnt[dst[e]], 1.0f);
    __syncthreads();
    float* dv = dinv + (size_t)t * N_NODES;
    for (int i = tid; i < N_NODES; i += 256) dv[i] = rsqrtf(cnt[i]);
}

// ---------------------------------------------------------------------------
// Kernel 2: H[T*N, 128] = X[T*N, 256] @ W_gcn[256, 128] via v_wmma_f32_16x16x32_f16.
//
// Block = 256 threads = 8 wave32s, each wave computes a 16(M) x 128(N) slab.
// W_gcn is converted f32->f16 and stored in LDS pre-swizzled into WMMA
// B-fragment order: Bs[tile(ks8*8+nt)][lane][elem], 32 B contiguous per lane.
//
// A-fragment 16-bit layout (ISA 7.12.2): lane l holds row M = l%16;
//   element e (0..15): K = (e/8)*16 + (l/16)*8 + (e%8).
// D-fragment f32 layout: lane l, vgpr v -> M = v + 8*(l/16), N = l%16.
// ---------------------------------------------------------------------------
__global__ __launch_bounds__(256)
void gcn_gemm_kernel(const float* __restrict__ X, const float* __restrict__ W,
                     float* __restrict__ H) {
    // 8 K-steps * 8 N-tiles = 64 tiles, 32 lanes * 16 halfs each = 64 KB LDS
    __shared__ _Float16 Bs[64 * 32 * 16];

    const int tid  = threadIdx.x;
    const int wave = tid >> 5;
    const int lane = tid & 31;
    const int lh   = lane >> 4;   // lane half (0/1)
    const int ln   = lane & 15;

    // --- pre-swizzle W_gcn into B-fragment order -------------------------
    for (int i = tid; i < 64 * 32 * 16; i += 256) {
        const int e     = i & 15;
        const int lane_ = (i >> 4) & 31;
        const int tile  = i >> 9;
        const int ks8   = tile >> 3;        // which K-step (0..7)
        const int nt    = tile & 7;         // which N-tile (0..7)
        const int kk    = ks8 * 32 + ((e >> 3) << 4) + ((lane_ >> 4) << 3) + (e & 7);
        const int col   = nt * 16 + (lane_ & 15);
        Bs[i] = (_Float16)W[kk * HID + col];
    }
    __syncthreads();
    const v16h* __restrict__ Bf = (const v16h*)Bs;

    v8f acc[8] = {};  // 8 N-tiles worth of 16x16 f32 accumulators

    const long aRow = (long)blockIdx.x * 128 + wave * 16 + ln; // A row this lane feeds
    const float* __restrict__ arow = X + aRow * NODE_DIM;

    for (int ks8 = 0; ks8 < 8; ++ks8) {
        const int ks = ks8 * 32;
        if (ks8 < 7) __builtin_prefetch(arow + ks + 32 + lh * 8, 0, 1); // global_prefetch_b8

        // Build A fragment: two groups of 8 contiguous f32 -> f16
        v16h afrag;
#pragma unroll
        for (int g = 0; g < 2; ++g) {
            const float4 lo = *(const float4*)(arow + ks + g * 16 + lh * 8);
            const float4 hi = *(const float4*)(arow + ks + g * 16 + lh * 8 + 4);
            afrag[g * 8 + 0] = (_Float16)lo.x;
            afrag[g * 8 + 1] = (_Float16)lo.y;
            afrag[g * 8 + 2] = (_Float16)lo.z;
            afrag[g * 8 + 3] = (_Float16)lo.w;
            afrag[g * 8 + 4] = (_Float16)hi.x;
            afrag[g * 8 + 5] = (_Float16)hi.y;
            afrag[g * 8 + 6] = (_Float16)hi.z;
            afrag[g * 8 + 7] = (_Float16)hi.w;
        }

#pragma unroll
        for (int nt = 0; nt < 8; ++nt) {
            const v16h bfrag = Bf[(ks8 * 8 + nt) * 32 + lane];
            acc[nt] = __builtin_amdgcn_wmma_f32_16x16x32_f16(
                /*neg_a=*/false, afrag, /*neg_b=*/false, bfrag,
                /*c_mod=*/(short)0, acc[nt], /*reuse_a=*/false, /*reuse_b=*/false);
        }
    }

    // --- store D fragments ----------------------------------------------
    const long outBase = (long)blockIdx.x * 128 + wave * 16 + 8 * lh;
#pragma unroll
    for (int nt = 0; nt < 8; ++nt) {
#pragma unroll
        for (int v = 0; v < 8; ++v) {
            H[(outBase + v) * HID + nt * 16 + ln] = acc[nt][v];
        }
    }
}

// ---------------------------------------------------------------------------
// Kernel 3: zero the aggregation buffer (float4 grid-stride-less exact cover)
// ---------------------------------------------------------------------------
__global__ __launch_bounds__(256)
void zero_kernel(float4* __restrict__ p) {
    const size_t i = (size_t)blockIdx.x * 256 + threadIdx.x;
    p[i] = make_float4(0.f, 0.f, 0.f, 0.f);
}

// ---------------------------------------------------------------------------
// Kernel 4: normalized scatter-add aggregation (edges + self loops).
// One warp per edge; lane f handles features [4f, 4f+3]. h/agg tiles are
// ~1 MB per timestep -> the random traffic stays in the 192 MB L2.
// ---------------------------------------------------------------------------
__global__ __launch_bounds__(256)
void agg_kernel(const int* __restrict__ ei, const float* __restrict__ dinv,
                const float* __restrict__ H, float* __restrict__ agg) {
    const int t    = blockIdx.y;
    const int we   = blockIdx.x * 8 + (threadIdx.x >> 5);   // edge (or self-loop) id
    const int lane = threadIdx.x & 31;
    if (we >= N_EDGES + N_NODES) return;

    const int*   eit = ei   + (size_t)t * 2 * N_EDGES;
    const float* dv  = dinv + (size_t)t * N_NODES;

    int src, dst;
    float norm;
    if (we < N_EDGES) {
        src  = eit[we];
        dst  = eit[N_EDGES + we];
        norm = dv[src] * dv[dst];
    } else {
        src = dst = we - N_EDGES;
        const float d = dv[src];
        norm = d * d;
    }

    const float4 hv = *(const float4*)(H + ((size_t)t * N_NODES + src) * HID + lane * 4);
    float* ap = agg + ((size_t)t * N_NODES + dst) * HID + lane * 4;
    atomicAdd(ap + 0, hv.x * norm);
    atomicAdd(ap + 1, hv.y * norm);
    atomicAdd(ap + 2, hv.z * norm);
    atomicAdd(ap + 3, hv.w * norm);
}

// ---------------------------------------------------------------------------
// Kernel 5: g[t] = mean_n tanh(agg[t][n] + b_gcn)   (128 threads, one per feat)
// ---------------------------------------------------------------------------
__global__ __launch_bounds__(128)
void pool_kernel(const float* __restrict__ agg, const float* __restrict__ b_gcn,
                 float* __restrict__ g) {
    const int t = blockIdx.x;
    const int f = threadIdx.x;
    const float b = b_gcn[f];
    float s = 0.f;
    const float* base = agg + (size_t)t * N_NODES * HID + f;
    for (int n = 0; n < N_NODES; ++n) s += tanhf(base[(size_t)n * HID] + b);
    g[t * HID + f] = s * (1.0f / N_NODES);
}

// ---------------------------------------------------------------------------
// Kernel 6: the sequential LSTM scan + head. One block, 128 threads.
// Thread j owns hidden unit j (gate rows j, 128+j, 256+j, 384+j).
// ---------------------------------------------------------------------------
__device__ __forceinline__ float sigmoidf(float x) { return 1.0f / (1.0f + expf(-x)); }

__global__ __launch_bounds__(128)
void lstm_kernel(const float* __restrict__ g, const float* __restrict__ kpi,
                 const float* __restrict__ W_ih, const float* __restrict__ W_hh,
                 const float* __restrict__ b_ih, const float* __restrict__ b_hh,
                 const float* __restrict__ W_head, const float* __restrict__ b_head,
                 float* __restrict__ out) {
    __shared__ float xs[IN_DIM];
    __shared__ float hs[HID];
    __shared__ float red[HID];

    const int j = threadIdx.x;
    hs[j] = 0.f;
    float c = 0.f;

    const float bi = b_ih[j]       + b_hh[j];
    const float bf = b_ih[128 + j] + b_hh[128 + j];
    const float bg = b_ih[256 + j] + b_hh[256 + j];
    const float bo = b_ih[384 + j] + b_hh[384 + j];

    const float* Wi_i = W_ih + (size_t)(j)       * IN_DIM;
    const float* Wi_f = W_ih + (size_t)(128 + j) * IN_DIM;
    const float* Wi_g = W_ih + (size_t)(256 + j) * IN_DIM;
    const float* Wi_o = W_ih + (size_t)(384 + j) * IN_DIM;
    const float* Wh_i = W_hh + (size_t)(j)       * HID;
    const float* Wh_f = W_hh + (size_t)(128 + j) * HID;
    const float* Wh_g = W_hh + (size_t)(256 + j) * HID;
    const float* Wh_o = W_hh + (size_t)(384 + j) * HID;

    for (int t = 0; t < T_STEPS; ++t) {
        xs[j] = g[t * HID + j];
        if (j < KPI_DIM) xs[HID + j] = kpi[t * KPI_DIM + j];
        __syncthreads();

        float ai = bi, af = bf, ag = bg, ao = bo;
        for (int k = 0; k < IN_DIM; ++k) {
            const float x = xs[k];
            ai += Wi_i[k] * x;  af += Wi_f[k] * x;
            ag += Wi_g[k] * x;  ao += Wi_o[k] * x;
        }
        for (int k = 0; k < HID; ++k) {
            const float h = hs[k];
            ai += Wh_i[k] * h;  af += Wh_f[k] * h;
            ag += Wh_g[k] * h;  ao += Wh_o[k] * h;
        }
        const float ig = sigmoidf(ai);
        const float fg = sigmoidf(af);
        const float gg = tanhf(ag);
        const float og = sigmoidf(ao);
        c = fg * c + ig * gg;
        const float hn = og * tanhf(c);
        __syncthreads();
        hs[j] = hn;
    }
    __syncthreads();

    // head: out = h . W_head + b_head
    red[j] = hs[j] * W_head[j];
    __syncthreads();
    for (int s = 64; s > 0; s >>= 1) {
        if (j < s) red[j] += red[j + s];
        __syncthreads();
    }
    if (j == 0) out[0] = red[0] + b_head[0];
}

// ---------------------------------------------------------------------------
// Host entry
// ---------------------------------------------------------------------------
extern "C" void kernel_launch(void* const* d_in, const int* in_sizes, int n_in,
                              void* d_out, int out_size, void* d_ws, size_t ws_size,
                              hipStream_t stream) {
    (void)in_sizes; (void)n_in; (void)out_size; (void)ws_size;

    const float* x_nodes    = (const float*)d_in[0];
    const int*   edge_index = (const int*)  d_in[1];
    const float* kpi        = (const float*)d_in[2];
    const float* W_gcn      = (const float*)d_in[3];
    const float* b_gcn      = (const float*)d_in[4];
    const float* W_ih       = (const float*)d_in[5];
    const float* W_hh       = (const float*)d_in[6];
    const float* b_ih       = (const float*)d_in[7];
    const float* b_hh       = (const float*)d_in[8];
    const float* W_head     = (const float*)d_in[9];
    const float* b_head     = (const float*)d_in[10];
    float* out = (float*)d_out;

    // workspace carve-out
    const size_t DINV_BYTES = (size_t)T_STEPS * N_NODES * sizeof(float);          // 2 MB
    const size_t H_BYTES    = (size_t)T_STEPS * N_NODES * HID * sizeof(float);    // 262 MB
    char* ws = (char*)d_ws;
    float* dinv = (float*)ws;
    float* Hbuf = (float*)(ws + DINV_BYTES);
    float* agg  = (float*)(ws + DINV_BYTES + H_BYTES);
    float* gbuf = (float*)(ws + DINV_BYTES + 2 * H_BYTES);

    // 1) degrees / symmetric norm
    deg_kernel<<<T_STEPS, 256, 0, stream>>>(edge_index, dinv);

    // 2) big WMMA GEMM: H = X @ W_gcn   (512000 rows / 128 rows-per-block)
    gcn_gemm_kernel<<<M_ROWS / 128, 256, 0, stream>>>(x_nodes, W_gcn, Hbuf);

    // 3) zero agg, then 4) normalized scatter-add aggregation
    const size_t aggF4 = (size_t)T_STEPS * N_NODES * HID / 4;   // 16,384,000 float4
    zero_kernel<<<(unsigned)(aggF4 / 256), 256, 0, stream>>>((float4*)agg);
    agg_kernel<<<dim3((N_EDGES + N_NODES) / 8, T_STEPS), 256, 0, stream>>>(
        edge_index, dinv, Hbuf, agg);

    // 5) tanh + mean pool per timestep
    pool_kernel<<<T_STEPS, HID, 0, stream>>>(agg, b_gcn, gbuf);

    // 6) sequential LSTM + linear head
    lstm_kernel<<<1, HID, 0, stream>>>(gbuf, kpi, W_ih, W_hh, b_ih, b_hh,
                                       W_head, b_head, out);
}
